// GIConv2dWeights_27075473834356
// MI455X (gfx1250) — compile-verified, matrix operand order checked by reading, not run on hardware
//
#include <hip/hip_runtime.h>
#include <hip/hip_bf16.h>

// ---- problem constants (match reference) ----
#define S_    8
#define P_    32
#define CIN_  128
#define H_    16
#define W_    16
#define HW_   256
#define COUT_ 256
#define K_    3
#define NN    1152        // N = CIN*K*K
#define PHW   8192        // P*H*W  (GEMM K-dimension)
#define KD    8192

typedef __bf16 bf16;
typedef __attribute__((ext_vector_type(16))) __bf16 v16bf;
typedef __attribute__((ext_vector_type(8)))  float  v8f;

// float -> bf16 (round-to-nearest-even), storage-only use of __bf16
static __device__ inline bf16 f2bf(float f) {
    union { float f; unsigned u; } c; c.f = f;
    unsigned u = c.u + 0x7FFFu + ((c.u >> 16) & 1u);
    union { unsigned short s; bf16 b; } r;
    r.s = (unsigned short)(u >> 16);
    return r.b;
}

// -------------------------------------------------------------------------
// Kernel 1: scale + unfold X[s] -> Xu bf16 [NN, PHW], feature order (c,kh,kw)
// -------------------------------------------------------------------------
__global__ void k_unfold(const float* __restrict__ X,
                         const float* __restrict__ lps,
                         bf16* __restrict__ Xu, int s) {
    int gid = blockIdx.x * blockDim.x + threadIdx.x;
    if (gid >= NN * PHW) return;
    int n   = gid / PHW;
    int col = gid - n * PHW;
    int p = col >> 8;          // column = p*HW + l
    int l = col & 255;
    int y = l >> 4, x = l & 15;
    int c = n / 9;
    int r = n - c * 9;
    int kh = r / 3, kw = r - kh * 3;
    int yy = y + kh - 1, xx = x + kw - 1;
    float v = 0.f;
    if (yy >= 0 && yy < H_ && xx >= 0 && xx < W_) {
        float sp = expf(0.5f * lps[p]);
        v = sp * X[(((size_t)s * P_ + p) * CIN_ + c) * (size_t)HW_ + yy * W_ + xx];
    }
    Xu[(size_t)n * PHW + col] = f2bf(v);
}

// -------------------------------------------------------------------------
// Kernel 2: scale u -> Yb bf16 [COUT, PHW], Yb[o, p*HW+l] = sp[p]*u[p,o,l]
// -------------------------------------------------------------------------
__global__ void k_scale_u(const float* __restrict__ u,
                          const float* __restrict__ lps,
                          bf16* __restrict__ Yb) {
    int gid = blockIdx.x * blockDim.x + threadIdx.x;
    if (gid >= COUT_ * PHW) return;
    int o   = gid / PHW;
    int col = gid - o * PHW;
    int p = col >> 8;
    int l = col & 255;
    float sp = expf(0.5f * lps[p]);
    Yb[(size_t)o * PHW + col] = f2bf(sp * u[((size_t)p * COUT_ + o) * HW_ + l]);
}

// -------------------------------------------------------------------------
// Kernel 3: C[M,Nc] = A[M,KD] * B[Nc,KD]^T, workgroup-cooperative WMMA GEMM.
//   Workgroup (8 waves) computes a 128x128 C tile. Per K-step of 32, the
//   128x32 bf16 slabs of A and B are staged into LDS with
//   global_load_async_to_lds_b128 (double-buffered, ASYNCcnt-tracked),
//   then each wave computes a 64x32 subtile with v_wmma_f32_16x16x32_bf16.
// -------------------------------------------------------------------------
static __device__ inline v16bf lds_frag(const bf16* __restrict__ p) {
    union { uint4 u[2]; v16bf v; } cv;
    cv.u[0] = *reinterpret_cast<const uint4*>(p);        // K k..k+7
    cv.u[1] = *reinterpret_cast<const uint4*>(p + 16);   // K k+16..k+23
    return cv.v;
}

// copy one 128x32 bf16 slab (8KB = 512 x 16B chunks) global -> LDS, async
static __device__ inline void async_copy_slab(const bf16* __restrict__ gbase,
                                              unsigned ldsbase,
                                              int rowBase, int kk) {
    const int tid = threadIdx.x;
#pragma unroll
    for (int h = 0; h < 2; ++h) {
        int c = tid + h * 256;          // chunk id 0..511
        int r = c >> 2;                 // slab row 0..127
        int q = c & 3;                  // 16B chunk within row
        unsigned lds = ldsbase + (unsigned)(r * 64 + q * 16);
        unsigned long long ga = (unsigned long long)(size_t)
            ((const char*)gbase + ((size_t)(rowBase + r) * KD + kk) * 2 + q * 16);
        asm volatile("global_load_async_to_lds_b128 %0, %1, off"
                     :: "v"(lds), "v"(ga) : "memory");
    }
}

__global__ void __launch_bounds__(256)
k_gemm_abt(const bf16* __restrict__ A,
           const bf16* __restrict__ Bm,
           float* __restrict__ C,
           int M, int Nc, float diagAdd) {
    __shared__ bf16 smA[2][128 * 32];   // 2 x 8KB
    __shared__ bf16 smB[2][128 * 32];   // 2 x 8KB

    const int tilesN = Nc >> 7;                      // 128-wide tiles
    const int tm = blockIdx.x / tilesN;
    const int tn = blockIdx.x - tm * tilesN;
    const int tmBase = tm * 128;
    const int tnBase = tn * 128;

    const int lane = threadIdx.x & 31;
    const int wid  = threadIdx.x >> 5;               // 0..7
    const int wr   = wid >> 2;                       // 0..1 -> 64-row block
    const int wc   = wid & 3;                        // 0..3 -> 32-col block
    const int lr   = lane & 15;
    const int sel  = lane >> 4;

    const unsigned ldsA[2] = { (unsigned)(size_t)&smA[0][0],
                               (unsigned)(size_t)&smA[1][0] };
    const unsigned ldsB[2] = { (unsigned)(size_t)&smB[0][0],
                               (unsigned)(size_t)&smB[1][0] };

    v8f acc[4][2] = {};

    // prologue: stage K-slab 0
    async_copy_slab(A,  ldsA[0], tmBase, 0);
    async_copy_slab(Bm, ldsB[0], tnBase, 0);
    asm volatile("s_wait_asynccnt 0x0" ::: "memory");
    __syncthreads();

    for (int kk = 0; kk < KD; kk += 32) {
        const int cur = (kk >> 5) & 1;
        if (kk + 32 < KD) {             // stage next slab into other buffer
            async_copy_slab(A,  ldsA[1 - cur], tmBase, kk + 32);
            async_copy_slab(Bm, ldsB[1 - cur], tnBase, kk + 32);
            // warm L2 two slabs ahead
            __builtin_prefetch(A  + (size_t)(tmBase + (threadIdx.x >> 1)) * KD + kk + 64, 0, 1);
            __builtin_prefetch(Bm + (size_t)(tnBase + (threadIdx.x >> 1)) * KD + kk + 64, 0, 1);
        }

        // fragments from LDS (ds_load_b128 pairs)
        v16bf af[4], bf_[2];
#pragma unroll
        for (int i = 0; i < 4; ++i)
            af[i] = lds_frag(&smA[cur][(wr * 64 + i * 16 + lr) * 32 + sel * 8]);
#pragma unroll
        for (int j = 0; j < 2; ++j)
            bf_[j] = lds_frag(&smB[cur][(wc * 32 + j * 16 + lr) * 32 + sel * 8]);

#pragma unroll
        for (int i = 0; i < 4; ++i)
#pragma unroll
            for (int j = 0; j < 2; ++j)
                acc[i][j] = __builtin_amdgcn_wmma_f32_16x16x32_bf16(
                    false, af[i], false, bf_[j], (short)0, acc[i][j], false, false);

        asm volatile("s_wait_asynccnt 0x0" ::: "memory");
        __syncthreads();
    }

    // C/D layout: VGPR v -> row = v + sel*8 ; col = lr
#pragma unroll
    for (int i = 0; i < 4; ++i) {
#pragma unroll
        for (int j = 0; j < 2; ++j) {
            const int baseRow = tmBase + wr * 64 + i * 16 + sel * 8;
            const int col     = tnBase + wc * 32 + j * 16 + lr;
#pragma unroll
            for (int v = 0; v < 8; ++v) {
                int row = baseRow + v;
                float val = acc[i][j][v];
                if (diagAdd != 0.f && row == col) val += diagAdd;
                C[(size_t)row * Nc + col] = val;
            }
        }
    }
}

// -------------------------------------------------------------------------
// Kernel 4: in-place right-looking Cholesky of A [NN,NN] (lower), + logdet
// -------------------------------------------------------------------------
__global__ void k_cholesky(float* __restrict__ A, float* __restrict__ logdet) {
    const int tid = threadIdx.x;
    const int nt  = blockDim.x;
    for (int j = 0; j < NN; ++j) {
        if (tid == 0) A[(size_t)j * NN + j] = sqrtf(A[(size_t)j * NN + j]);
        __syncthreads();
        float d = A[(size_t)j * NN + j];
        for (int i = j + 1 + tid; i < NN; i += nt)
            A[(size_t)i * NN + j] /= d;
        __syncthreads();
        for (int i = j + 1 + tid; i < NN; i += nt) {
            float lij = A[(size_t)i * NN + j];
            float* rowi = A + (size_t)i * NN;
            for (int m = j + 1; m <= i; ++m)
                rowi[m] -= lij * A[(size_t)m * NN + j];
        }
        __syncthreads();
    }
    __shared__ float red[1024];
    float part = 0.f;
    for (int j = tid; j < NN; j += nt)
        part += logf(A[(size_t)j * NN + j]);
    red[tid] = part;
    __syncthreads();
    for (int off = nt >> 1; off > 0; off >>= 1) {
        if (tid < off) red[tid] += red[tid + off];
        __syncthreads();
    }
    if (tid == 0) *logdet = 2.f * red[0];
}

// -------------------------------------------------------------------------
// Kernel 5: triangular solves. Thread o (= output channel) owns one column.
// -------------------------------------------------------------------------
__global__ void k_solve(const float* __restrict__ L,
                        const float* __restrict__ Bxly,   // [NN, COUT]
                        const float* __restrict__ Zs,     // [COUT, NN]
                        float* __restrict__ ybuf,         // [NN, COUT]
                        float* __restrict__ out) {        // [COUT, NN]
    const int o = threadIdx.x;
    // forward substitution: L y = XLY
    for (int i = 0; i < NN; ++i) {
        float sum = Bxly[(size_t)i * COUT_ + o];
        const float* Lrow = L + (size_t)i * NN;
        for (int k = 0; k < i; ++k)
            sum -= Lrow[k] * ybuf[(size_t)k * COUT_ + o];
        ybuf[(size_t)i * COUT_ + o] = sum / Lrow[i];
    }
    // backward: L^T mean = y (in place)
    for (int i = NN - 1; i >= 0; --i) {
        float sum = ybuf[(size_t)i * COUT_ + o];
        for (int k = i + 1; k < NN; ++k)
            sum -= L[(size_t)k * NN + i] * ybuf[(size_t)k * COUT_ + o];
        ybuf[(size_t)i * COUT_ + o] = sum / L[(size_t)i * NN + i];
    }
    // backward: L^T dW = Z column, stored directly in out[o,:]
    float*       wrow = out + (size_t)o * NN;
    const float* zrow = Zs  + (size_t)o * NN;
    for (int i = NN - 1; i >= 0; --i) {
        float sum = zrow[i];
        for (int k = i + 1; k < NN; ++k)
            sum -= L[(size_t)k * NN + i] * wrow[k];
        wrow[i] = sum / L[(size_t)i * NN + i];
    }
    // sample = mean + dW
    for (int i = 0; i < NN; ++i)
        wrow[i] += ybuf[(size_t)i * COUT_ + o];
}

// -------------------------------------------------------------------------
// Kernel 6: logpq[s] = logP - logQ
// -------------------------------------------------------------------------
__global__ void k_logpq(const float* __restrict__ sample,  // [COUT*NN]
                        const float* __restrict__ Zs,      // [COUT*NN]
                        const float* __restrict__ logdet,
                        float* __restrict__ outv) {
    __shared__ float r1[256], r2[256];
    const int tid = threadIdx.x;
    float a = 0.f, b = 0.f;
    for (int i = tid; i < COUT_ * NN; i += 256) {
        float sv = sample[i]; a += sv * sv;
        float zv = Zs[i];     b += zv * zv;
    }
    r1[tid] = a; r2[tid] = b;
    __syncthreads();
    for (int off = 128; off > 0; off >>= 1) {
        if (tid < off) { r1[tid] += r1[tid + off]; r2[tid] += r2[tid + off]; }
        __syncthreads();
    }
    if (tid == 0) {
        const float fan_in = (float)NN;
        float logP = 0.5f * COUT_ * NN * logf(fan_in) - 0.5f * fan_in * r1[0];
        float logQ = -0.5f * r2[0] + 0.5f * (float)COUT_ * (*logdet);
        *outv = logP - logQ;
    }
}

// -------------------------------------------------------------------------
extern "C" void kernel_launch(void* const* d_in, const int* in_sizes, int n_in,
                              void* d_out, int out_size, void* d_ws, size_t ws_size,
                              hipStream_t stream) {
    const float* X   = (const float*)d_in[0];   // [S,P,CIN,H,W]
    const float* u   = (const float*)d_in[1];   // [P,COUT,H,W]
    const float* lps = (const float*)d_in[2];   // [P]
    const float* Z   = (const float*)d_in[3];   // [S,COUT,NN]
    float* out = (float*)d_out;                 // sample [S,COUT,NN] then logpq [S]
    char* ws = (char*)d_ws;

    // workspace layout (256B aligned offsets)
    bf16*  Xu     = (bf16*) (ws + 0);            // NN*PHW bf16  = 18,874,368 B
    bf16*  Yb     = (bf16*) (ws + 18874368);     // COUT*PHW bf16=  4,194,304 B
    float* prec   = (float*)(ws + 23068672);     // NN*NN f32    =  5,308,416 B
    float* XLY    = (float*)(ws + 28377088);     // NN*COUT f32  =  1,179,648 B
    float* ybuf   = (float*)(ws + 29556736);     // NN*COUT f32  =  1,179,648 B
    float* logdet = (float*)(ws + 30736384);     // S f32

    dim3 b256(256);

    // Yb once (shared across s)
    k_scale_u<<<(COUT_ * PHW + 255) / 256, b256, 0, stream>>>(u, lps, Yb);

    const int blocksXLX = (NN / 128) * (NN / 128);     // 9*9 = 81 tiles
    const int blocksXLY = (NN / 128) * (COUT_ / 128);  // 9*2 = 18 tiles

    for (int s = 0; s < S_; ++s) {
        k_unfold<<<(NN * PHW + 255) / 256, b256, 0, stream>>>(X, lps, Xu, s);

        k_gemm_abt<<<blocksXLX, b256, 0, stream>>>(
            Xu, Xu, prec, NN, NN, (float)NN /* fan_in * I */);

        k_gemm_abt<<<blocksXLY, b256, 0, stream>>>(
            Xu, Yb, XLY, NN, COUT_, 0.f);

        k_cholesky<<<1, 1024, 0, stream>>>(prec, logdet + s);

        k_solve<<<1, 256, 0, stream>>>(
            prec, XLY, Z + (size_t)s * COUT_ * NN, ybuf,
            out + (size_t)s * COUT_ * NN);

        k_logpq<<<1, 256, 0, stream>>>(
            out + (size_t)s * COUT_ * NN, Z + (size_t)s * COUT_ * NN,
            logdet + s, out + (size_t)S_ * COUT_ * NN + s);
    }
}
